// LinearRNN_2576980378048
// MI455X (gfx1250) — matrix operand order repaired and checked by reading, not compiled, not run on hardware
//
#include <hip/hip_runtime.h>

// CDNA5 / gfx1250: wave32, v_wmma_f32_16x16x32_bf16.
// LinearRNN y_t = A y_{t-1}, A = I + 1e-4*N, D=4096, 4096 steps.
//
// W = A - I; squaring W' = W^2 + 2W gives (I+W)' = (I+W)^2, so 6 squarings
// yield W64 = A^64 - I with the identity factored out exactly (bf16-safe).
// All matrix operands are compensated bf16 hi/lo pairs; each logical product
// is 3 WMMAs (hi*hi + hi*lo + lo*hi) accumulated in f32 (~2^-17 precision).
//
// Schedule: split -> 4 squarings (W16) -> 16 GEMVs -> 3 N=16 block GEMMs
// (fill rows 16..63 with W16) -> 2 more squarings (W64) -> 63 block GEMMs
// Y_b = Y_{b-1} + W64 @ Y_{b-1}, streamed straight into d_out (time-major
// Y layout == output layout == NT-GEMM B-operand layout).

typedef __attribute__((ext_vector_type(16))) __bf16 v16bf;
typedef __attribute__((ext_vector_type(8)))  __bf16 v8bf;
typedef __attribute__((ext_vector_type(8)))  float  v8f;

#define D_DIM 4096
#define STEPS 4096
#define BLK   64      // time-block size for W64 application
#define SEED  16      // rows seeded by sequential GEMV
#define NSQ   6       // log2(BLK)

union V16U { v16bf v; v8bf h[2]; };

static __device__ __forceinline__ void bf16_split(float x, __bf16& h, __bf16& l) {
  h = (__bf16)x;
  l = (__bf16)(x - (float)h);
}

// ---- build W1 = A - I as row-major + transposed compensated bf16 pairs ----
__global__ void __launch_bounds__(256)
split_kernel(const float* __restrict__ A,
             __bf16* __restrict__ Er_hi, __bf16* __restrict__ Er_lo,
             __bf16* __restrict__ Ec_hi, __bf16* __restrict__ Ec_lo) {
  size_t idx = (size_t)blockIdx.x * 256 + threadIdx.x;
  int i = (int)(idx >> 12);
  int j = (int)(idx & 4095);
  float e = A[idx] - (i == j ? 1.0f : 0.0f);
  __bf16 h, l; bf16_split(e, h, l);
  Er_hi[idx] = h; Er_lo[idx] = l;
  size_t t = ((size_t)j << 12) + (size_t)i;
  Ec_hi[t] = h; Ec_lo[t] = l;
}

// ---- f32 GEMV: y_out = A @ y_prev ; optionally emit bf16 hi/lo row ----
__global__ void __launch_bounds__(256)
gemv_kernel(const float* __restrict__ A, const float* __restrict__ yprev,
            float* __restrict__ yout,
            __bf16* __restrict__ yhi, __bf16* __restrict__ ylo) {
  int row  = blockIdx.x * 8 + (threadIdx.x >> 5);   // 8 waves / block
  int lane = threadIdx.x & 31;
  const float* a = A + ((size_t)row << 12);
  float s = 0.0f;
  for (int k = lane; k < D_DIM; k += 32) s += a[k] * yprev[k];
  for (int off = 16; off > 0; off >>= 1) s += __shfl_xor(s, off, 32);
  if (lane == 0) {
    yout[row] = s;
    if (yhi) { __bf16 h, l; bf16_split(s, h, l); yhi[row] = h; ylo[row] = l; }
  }
}

// ---- compensated bf16 NT-GEMM on WMMA: C = A @ Bt^T (K = 4096) ----
// Wave layout: WM x WN waves (WM*WN == 8), wave tile (MI*16) x (NI*16).
// B rows clamp to nmax-1 (EXEC stays all-ones for WMMA); epilogue stores
// are guarded by n < nmax.
// mode 0 (squaring):  Z = acc + 2*W(m,n); write Z row-major + transposed hi/lo
// mode 1 (block app): Y = acc + Cprev[n][m]; write f32 + bf16 hi/lo
template<int WM, int WN, int MI, int NI>
__global__ void __launch_bounds__(256)
wmma_gemm_nt(const __bf16* __restrict__ Ahi, const __bf16* __restrict__ Alo,
             const __bf16* __restrict__ Bhi, const __bf16* __restrict__ Blo,
             int mode, int nmax,
             const __bf16* __restrict__ Whi, const __bf16* __restrict__ Wlo,
             __bf16* __restrict__ Zr_hi, __bf16* __restrict__ Zr_lo,
             __bf16* __restrict__ Zc_hi, __bf16* __restrict__ Zc_lo,
             const float* __restrict__ Cprev, float* __restrict__ Cout,
             __bf16* __restrict__ Yhi_out, __bf16* __restrict__ Ylo_out) {
  const int lane = threadIdx.x & 31;
  const int wv   = threadIdx.x >> 5;
  const int wm   = wv % WM;
  const int wn   = wv / WM;
  const int m0   = blockIdx.x * (WM * MI * 16) + wm * (MI * 16);
  const int n0   = blockIdx.y * (WN * NI * 16) + wn * (NI * 16);
  const int lm   = lane & 15;
  const int lk8  = (lane >> 4) << 4;   // A-frag k offset: 0 / 16  (contiguous, matches B)
  const int lk16 = (lane >> 4) << 4;   // B-frag k offset: 0 / 16  (contiguous, matches A)

  size_t rowA[MI], rowB[NI];
#pragma unroll
  for (int mi = 0; mi < MI; ++mi)
    rowA[mi] = ((size_t)(m0 + mi * 16 + lm) << 12);
#pragma unroll
  for (int ni = 0; ni < NI; ++ni) {
    int n = n0 + ni * 16 + lm;
    if (n >= nmax) n = nmax - 1;        // clamp, don't predicate
    rowB[ni] = ((size_t)n << 12);
  }

  v8f acc[MI][NI] = {};

  for (int kb = 0; kb < D_DIM; kb += 32) {
    V16U a_hi[MI], a_lo[MI];
    v16bf b_hi[NI], b_lo[NI];
#pragma unroll
    for (int mi = 0; mi < MI; ++mi) {
      const __bf16* p = Ahi + rowA[mi] + kb + lk8;
      a_hi[mi].h[0] = *(const v8bf*)p; a_hi[mi].h[1] = *(const v8bf*)(p + 8);
      p = Alo + rowA[mi] + kb + lk8;
      a_lo[mi].h[0] = *(const v8bf*)p; a_lo[mi].h[1] = *(const v8bf*)(p + 8);
    }
#pragma unroll
    for (int ni = 0; ni < NI; ++ni) {
      b_hi[ni] = *(const v16bf*)(Bhi + rowB[ni] + kb + lk16);
      b_lo[ni] = *(const v16bf*)(Blo + rowB[ni] + kb + lk16);
    }
#pragma unroll
    for (int mi = 0; mi < MI; ++mi)
#pragma unroll
      for (int ni = 0; ni < NI; ++ni) {
        acc[mi][ni] = __builtin_amdgcn_wmma_f32_16x16x32_bf16(
            false, a_hi[mi].v, false, b_hi[ni], (short)0, acc[mi][ni], false, false);
        acc[mi][ni] = __builtin_amdgcn_wmma_f32_16x16x32_bf16(
            false, a_hi[mi].v, false, b_lo[ni], (short)0, acc[mi][ni], false, false);
        acc[mi][ni] = __builtin_amdgcn_wmma_f32_16x16x32_bf16(
            false, a_lo[mi].v, false, b_hi[ni], (short)0, acc[mi][ni], false, false);
      }
  }

  // epilogue: C/D layout — lane holds n = lm (per subtile), M = (lane>>4)*8 + r
  const int mbase = (lane >> 4) * 8;
#pragma unroll
  for (int mi = 0; mi < MI; ++mi)
#pragma unroll
    for (int ni = 0; ni < NI; ++ni) {
      const int n = n0 + ni * 16 + lm;
      if (n >= nmax) continue;          // divergence only after all WMMAs
#pragma unroll
      for (int r = 0; r < 8; ++r) {
        const int m = m0 + mi * 16 + mbase + r;
        float v = acc[mi][ni][r];
        if (mode == 0) {
          size_t idx = ((size_t)m << 12) + (size_t)n;
          v += 2.0f * ((float)Whi[idx] + (float)Wlo[idx]);   // W' = W^2 + 2W
          __bf16 h, l; bf16_split(v, h, l);
          Zr_hi[idx] = h; Zr_lo[idx] = l;
          size_t idt = ((size_t)n << 12) + (size_t)m;
          Zc_hi[idt] = h; Zc_lo[idt] = l;
        } else {
          size_t idx = ((size_t)n << 12) + (size_t)m;        // time-major
          v += Cprev[idx];                                   // Y = Yprev + W*Yprev
          Cout[idx] = v;
          __bf16 h, l; bf16_split(v, h, l);
          Yhi_out[idx] = h; Ylo_out[idx] = l;
        }
      }
    }
}

extern "C" void kernel_launch(void* const* d_in, const int* in_sizes, int n_in,
                              void* d_out, int out_size, void* d_ws, size_t ws_size,
                              hipStream_t stream) {
  (void)in_sizes; (void)n_in; (void)out_size;
  const float* A      = (const float*)d_in[1];   // (D, D) f32
  const float* y_init = (const float*)d_in[2];   // (2, D) f32; row 0 used
  float* out = (float*)d_out;                    // (STEPS, D) f32

  const size_t DD     = (size_t)D_DIM * D_DIM;
  const size_t wbytes = DD * sizeof(__bf16);                 // 32 MB each
  const size_t ybytes = (size_t)BLK * D_DIM * sizeof(__bf16);
  const size_t need   = 8 * wbytes + 4 * ybytes;             // ~258 MB

  if (ws_size < need) {
    // fallback: plain sequential GEMV chain (A stays L2-resident)
    for (int t = 0; t < STEPS; ++t)
      gemv_kernel<<<D_DIM / 8, 256, 0, stream>>>(
          A, t ? out + (size_t)(t - 1) * D_DIM : y_init,
          out + (size_t)t * D_DIM, nullptr, nullptr);
    return;
  }

  char* w = (char*)d_ws;
  __bf16 *Wr_hi[2], *Wr_lo[2], *Wc_hi[2], *Wc_lo[2];
  for (int p = 0; p < 2; ++p) {
    Wr_hi[p] = (__bf16*)w; w += wbytes;
    Wr_lo[p] = (__bf16*)w; w += wbytes;
    Wc_hi[p] = (__bf16*)w; w += wbytes;
    Wc_lo[p] = (__bf16*)w; w += wbytes;
  }
  __bf16 *Y_hi[2], *Y_lo[2];
  for (int p = 0; p < 2; ++p) {
    Y_hi[p] = (__bf16*)w; w += ybytes;
    Y_lo[p] = (__bf16*)w; w += ybytes;
  }

  // W1 = A - I (row-major + transposed, hi/lo)
  split_kernel<<<(int)(DD / 256), 256, 0, stream>>>(
      A, Wr_hi[0], Wr_lo[0], Wc_hi[0], Wc_lo[0]);

  // squarings 1..4: W16 = A^16 - I ends in set 0
  int src = 0;
  for (int s = 0; s < 4; ++s) {
    int dst = 1 - src;
    wmma_gemm_nt<4, 2, 2, 2><<<dim3(D_DIM / 128, D_DIM / 64), 256, 0, stream>>>(
        Wr_hi[src], Wr_lo[src], Wc_hi[src], Wc_lo[src], /*mode=*/0, D_DIM,
        Wr_hi[src], Wr_lo[src],
        Wr_hi[dst], Wr_lo[dst], Wc_hi[dst], Wc_lo[dst],
        nullptr, nullptr, nullptr, nullptr);
    src = dst;
  }

  // seed rows 0..15 sequentially (f32 GEMV), emitting bf16 Y rows
  for (int r = 0; r < SEED; ++r)
    gemv_kernel<<<D_DIM / 8, 256, 0, stream>>>(
        A, r ? out + (size_t)(r - 1) * D_DIM : y_init,
        out + (size_t)r * D_DIM,
        Y_hi[0] + (size_t)r * D_DIM, Y_lo[0] + (size_t)r * D_DIM);

  // rows 16..63 via 3 N=16 block GEMMs with W16 (reads set 0 only)
  for (int j = 1; j < BLK / SEED; ++j) {
    wmma_gemm_nt<8, 1, 1, 2><<<dim3(D_DIM / 128, 1), 256, 0, stream>>>(
        Wr_hi[src], Wr_lo[src],
        Y_hi[0] + (size_t)(j - 1) * SEED * D_DIM,
        Y_lo[0] + (size_t)(j - 1) * SEED * D_DIM, /*mode=*/1, SEED,
        nullptr, nullptr, nullptr, nullptr, nullptr, nullptr,
        out + (size_t)(j - 1) * SEED * D_DIM, out + (size_t)j * SEED * D_DIM,
        Y_hi[0] + (size_t)j * SEED * D_DIM, Y_lo[0] + (size_t)j * SEED * D_DIM);
  }

  // squarings 5..6: W64 = A^64 - I ends back in set 0
  for (int s = 0; s < 2; ++s) {
    int dst = 1 - src;
    wmma_gemm_nt<4, 2, 2, 2><<<dim3(D_DIM / 128, D_DIM / 64), 256, 0, stream>>>(
        Wr_hi[src], Wr_lo[src], Wc_hi[src], Wc_lo[src], /*mode=*/0, D_DIM,
        Wr_hi[src], Wr_lo[src],
        Wr_hi[dst], Wr_lo[dst], Wc_hi[dst], Wc_lo[dst],
        nullptr, nullptr, nullptr, nullptr);
    src = dst;
  }

  // 63 block GEMMs: Y_b = Y_{b-1} + W64 @ Y_{b-1}, streamed into d_out.
  // 8x1 wave layout -> grid (32,2): 64 WGs / 512 waves per launch.
  int yb = 0;
  for (int b = 1; b < STEPS / BLK; ++b) {
    wmma_gemm_nt<8, 1, 1, 2><<<dim3(D_DIM / 128, 2), 256, 0, stream>>>(
        Wr_hi[src], Wr_lo[src], Y_hi[yb], Y_lo[yb], /*mode=*/1, BLK,
        nullptr, nullptr, nullptr, nullptr, nullptr, nullptr,
        out + (size_t)(b - 1) * BLK * D_DIM, out + (size_t)b * BLK * D_DIM,
        Y_hi[1 - yb], Y_lo[1 - yb]);
    yb = 1 - yb;
  }
}
